// MPNN_28561532518785
// MI455X (gfx1250) — matrix operand order repaired and checked by reading, not compiled, not run on hardware
//
#include <hip/hip_runtime.h>

// ---------------------------------------------------------------------------
// Types for CDNA5 WMMA (wave32)
// ---------------------------------------------------------------------------
typedef __attribute__((ext_vector_type(16))) __bf16         bfx16;
typedef __attribute__((ext_vector_type(8)))  float          fx8;
typedef __attribute__((ext_vector_type(8)))  unsigned short usx8;
typedef __attribute__((ext_vector_type(4)))  unsigned int   u32x4;
typedef __attribute__((ext_vector_type(8)))  int            i32x8;
typedef __attribute__((ext_vector_type(4)))  int            i32x4;

union FragBF { bfx16 v; usx8 u[2]; };

#if defined(__has_builtin)
#if __has_builtin(__builtin_amdgcn_tensor_load_to_lds) && __has_builtin(__builtin_amdgcn_s_wait_tensorcnt)
#define HAVE_TDM 1
#endif
#endif
#ifndef HAVE_TDM
#define HAVE_TDM 0
#endif

__device__ __forceinline__ unsigned short f2bf(float f) {
  unsigned int u = __float_as_uint(f);
  u += 0x7FFFu + ((u >> 16) & 1u);   // round-to-nearest-even
  return (unsigned short)(u >> 16);
}
__device__ __forceinline__ float bf2f(unsigned short h) {
  return __uint_as_float((unsigned int)h << 16);
}
__device__ __forceinline__ float silu_f(float v) {
  return v * (1.0f / (1.0f + __expf(-v)));
}

// ---------------------------------------------------------------------------
// TDM: stage 64 rows x K bf16 of Bt[N,K] (rows n0..n0+63) into LDS.
// D# per CDNA5 ISA ch.8: group0 = {count|flags, lds_addr, gaddr_lo,
// gaddr_hi|type=2}; group1 = 2-D tile descriptor, data_size=1 (2 bytes).
// ---------------------------------------------------------------------------
__device__ __forceinline__ void stage_b_panel(const unsigned short* Bt,
                                              int n0, int K, int N,
                                              unsigned short* lds) {
#if HAVE_TDM
  unsigned long long ga = (unsigned long long)(const void*)(Bt + (size_t)n0 * K);
  unsigned int ldsoff = (unsigned int)(size_t)lds;   // LDS flat addr low 32 = LDS offset
  u32x4 g0;
  g0[0] = 1u;                                        // count=1, user mode
  g0[1] = ldsoff;                                    // lds_addr
  g0[2] = (unsigned int)ga;                          // global_addr[31:0]
  g0[3] = (unsigned int)((ga >> 32) & 0x01FFFFFFu)   // global_addr[56:32]
        | (2u << 30);                                // type=2 ("image")
  i32x8 g1;
  g1[0] = (int)(1u << 16);                           // wg_mask=0, data_size=1 (2B)
  g1[1] = (int)(((unsigned)K & 0xFFFFu) << 16);      // tensor_dim0[15:0]
  g1[2] = (int)((((unsigned)K >> 16) & 0xFFFFu)      // tensor_dim0[31:16]
        | (((unsigned)N & 0xFFFFu) << 16));          // tensor_dim1[15:0]
  g1[3] = (int)(((((unsigned)N >> 16) & 0xFFFFu))    // tensor_dim1[31:16]
        | (((unsigned)K & 0xFFFFu) << 16));          // tile_dim0 = K
  g1[4] = 64;                                        // tile_dim1 = 64 rows
  g1[5] = K;                                         // tensor_dim0_stride[31:0]
  g1[6] = 0;
  g1[7] = 0;
  i32x4 z4 = {0, 0, 0, 0};
#if __clang_major__ >= 23
  i32x8 z8 = {0, 0, 0, 0, 0, 0, 0, 0};
  __builtin_amdgcn_tensor_load_to_lds(g0, g1, z4, z4, z8, 0);
#else
  __builtin_amdgcn_tensor_load_to_lds(g0, g1, z4, z4, 0);
#endif
  __builtin_amdgcn_s_wait_tensorcnt(0);
#else
  // Fallback: cooperative wave32 copy (compiler inserts the needed waits)
  const int lane = threadIdx.x & 31;
  for (int r = lane; r < 64; r += 32) {
    const usx8* src = (const usx8*)(Bt + (size_t)(n0 + r) * K);
    usx8* dst = (usx8*)(lds + (size_t)r * K);
    for (int c = 0; c < K / 8; ++c) dst[c] = src[c];
  }
#endif
}

// ---------------------------------------------------------------------------
// WMMA GEMM:  C[M,N] = act( A[M,K](bf16) * B[K,N] + bias[N] )
// B pre-transposed as Bt[N,K] (bf16), 64-row panel staged in LDS via TDM.
// One wave -> 16x64 tile.  K is a compile-time constant so the k-loop fully
// unrolls: no loop-carried PHIs / register rotation; the scheduler hoists
// global A loads and LDS B loads across the straight-line WMMA stream.
//   A fragment (16x32 bf16): lane = 16*h + m; elems 0-7 : K = kb+8h+e
//                                            elems 8-15: K = kb+16+8h+(e-8)
//   B fragment (32x16 bf16): lane = 16*h + n; same K mapping as A
// Requires M%16==0, N%64==0 (true for all shapes here).
// ---------------------------------------------------------------------------
template<int ACT, int OUT_BF16, int KK>
__launch_bounds__(32)
__global__ void gemm_wmma_kernel(const unsigned short* __restrict__ A,
                                 const unsigned short* __restrict__ Bt,
                                 const float* __restrict__ bias,
                                 void* __restrict__ C,
                                 int M, int N)
{
  extern __shared__ unsigned short bpanel[];   // [64][KK] bf16

  const int lane = threadIdx.x & 31;
  const int h    = lane >> 4;
  const int l16  = lane & 15;
  const int m0   = blockIdx.y << 4;
  const int n0   = blockIdx.x << 6;

  stage_b_panel(Bt, n0, KK, N, bpanel);

  fx8 acc[4];
#pragma unroll
  for (int t = 0; t < 4; ++t)
#pragma unroll
    for (int r = 0; r < 8; ++r) acc[t][r] = 0.0f;

  const unsigned short* ap = A + (size_t)(m0 + l16) * (size_t)KK + 8 * h;

#pragma unroll
  for (int kb = 0; kb < KK; kb += 32) {
    FragBF a;
    a.u[0] = *(const usx8*)(ap + kb);
    a.u[1] = *(const usx8*)(ap + kb + 16);
#pragma unroll
    for (int t = 0; t < 4; ++t) {
      const unsigned short* brow =
          bpanel + (size_t)(16 * t + l16) * KK + kb + 8 * h;
      FragBF b;
      b.u[0] = *(const usx8*)(brow);
      b.u[1] = *(const usx8*)(brow + 16);
      acc[t] = __builtin_amdgcn_wmma_f32_16x16x32_bf16(
          false, a.v, false, b.v, (short)0, acc[t], false, false);
    }
  }

#pragma unroll
  for (int t = 0; t < 4; ++t) {
    const int n = n0 + 16 * t + l16;
    const float bv = bias[n];
#pragma unroll
    for (int r = 0; r < 8; ++r) {
      const int m = m0 + r + 8 * h;
      float v = acc[t][r] + bv;
      if (ACT) v = silu_f(v);
      if (OUT_BF16)
        ((unsigned short*)C)[(size_t)m * N + n] = f2bf(v);
      else
        ((float*)C)[(size_t)m * N + n] = v;
    }
  }
}

// ---------------------------------------------------------------------------
// Small helper kernels
// ---------------------------------------------------------------------------
__global__ void f32_to_bf16_kernel(const float* __restrict__ x,
                                   unsigned short* __restrict__ y, int n) {
  int i = blockIdx.x * 256 + threadIdx.x;
  if (i < n) y[i] = f2bf(x[i]);
}

// W[K,N] fp32 -> Wt[N,K] bf16
__global__ void transpose_w_kernel(const float* __restrict__ W,
                                   unsigned short* __restrict__ Wt,
                                   int Kd, int Nd) {
  int i = blockIdx.x * 256 + threadIdx.x;
  if (i >= Kd * Nd) return;
  int k = i / Nd, n = i - k * Nd;
  Wt[(size_t)n * Kd + k] = f2bf(W[i]);
}

__device__ __forceinline__ float block_sum_128(float v, float* s) {
  int t = threadIdx.x;
  s[t] = v; __syncthreads();
  if (t < 64) s[t] += s[t + 64]; __syncthreads();
  if (t < 32) s[t] += s[t + 32]; __syncthreads();
  if (t < 16) s[t] += s[t + 16]; __syncthreads();
  if (t <  8) s[t] += s[t +  8]; __syncthreads();
  if (t <  4) s[t] += s[t +  4]; __syncthreads();
  if (t <  2) s[t] += s[t +  2]; __syncthreads();
  if (t <  1) s[t] += s[t +  1]; __syncthreads();
  float r = s[0]; __syncthreads();
  return r;
}

// One block (128 thr) per edge: gather [Vi|Vj|E] (384), unbiased LN,
// per-node gamma/beta affine -> bf16 row of X [M3,384].
__global__ void prelude_kernel(const float* __restrict__ V,
                               const float* __restrict__ E,
                               const int*   __restrict__ Kidx,
                               const float* __restrict__ gb,   // [ZN,768]
                               unsigned short* __restrict__ X)
{
  __shared__ float sbuf[128];
  const int e  = blockIdx.x;          // edge index
  const int zn = e / 48;
  const int z  = zn >> 11;
  const int n  = zn & 2047;
  const int j  = Kidx[e];
  const int d  = threadIdx.x;

  const float* vi = V + ((size_t)z * 2048 + n) * 128;
  const float* vj = V + ((size_t)z * 2048 + j) * 128;
  const float* ee = E + (size_t)e * 128;

  float x0 = vi[d], x1 = vj[d], x2 = ee[d];
  float mu = block_sum_128(x0 + x1 + x2, sbuf) * (1.0f / 384.0f);
  float c0 = x0 - mu, c1 = x1 - mu, c2 = x2 - mu;
  float var = block_sum_128(c0 * c0 + c1 * c1 + c2 * c2, sbuf) * (1.0f / 383.0f);
  float st  = sqrtf(var);
  float inv = (st == 0.0f) ? 1.0f : 1.0f / st;

  const float* g = gb + (size_t)zn * 768;
  const float* b = g + 384;
  unsigned short* xo = X + (size_t)e * 384;
  xo[d      ] = f2bf(g[d      ] * (c0 * inv) + b[d      ]);
  xo[d + 128] = f2bf(g[d + 128] * (c1 * inv) + b[d + 128]);
  xo[d + 256] = f2bf(g[d + 256] * (c2 * inv) + b[d + 256]);
}

// One block (128 thr) per node: Mv = a1 * sum_k(msg*m); x = V+Mv; LN; g2,b2.
__global__ void node_update_kernel(const unsigned short* __restrict__ MSG, // [ZN,48,128] bf16
                                   const float* __restrict__ mask,         // [ZN,48]
                                   const float* __restrict__ a1,           // [ZN,128]
                                   const float* __restrict__ V,            // [ZN,128]
                                   const float* __restrict__ gb2,          // [ZN,256]
                                   float* __restrict__ Vln,
                                   unsigned short* __restrict__ Vln_bf)
{
  __shared__ float sbuf[128];
  const int zn = blockIdx.x;
  const int d  = threadIdx.x;
  const unsigned short* mrow = MSG + (size_t)zn * 48 * 128 + d;
  const float* mk = mask + (size_t)zn * 48;
  float s = 0.0f;
#pragma unroll 4
  for (int k = 0; k < 48; ++k) s += bf2f(mrow[(size_t)k * 128]) * mk[k];

  float x = V[(size_t)zn * 128 + d] + a1[(size_t)zn * 128 + d] * s;
  float mu = block_sum_128(x, sbuf) * (1.0f / 128.0f);
  float c  = x - mu;
  float var = block_sum_128(c * c, sbuf) * (1.0f / 127.0f);
  float st  = sqrtf(var);
  float inv = (st == 0.0f) ? 1.0f : 1.0f / st;
  float y = gb2[(size_t)zn * 256 + d] * (c * inv) + gb2[(size_t)zn * 256 + 128 + d];
  Vln[(size_t)zn * 128 + d]    = y;
  Vln_bf[(size_t)zn * 128 + d] = f2bf(y);
}

__global__ void v2_kernel(const float* __restrict__ Vln,
                          const float* __restrict__ a2,
                          const float* __restrict__ FO,
                          float* __restrict__ V2,
                          float* __restrict__ Vout, int n) {
  int i = blockIdx.x * 256 + threadIdx.x;
  if (i >= n) return;
  float v = Vln[i] + a2[i] * FO[i];
  V2[i] = v;
  Vout[i] = v;
}

__global__ void e_update_kernel(const float* __restrict__ E,
                                const float* __restrict__ a3,   // [ZN,128]
                                const unsigned short* __restrict__ MSGE,
                                const float* __restrict__ mask, // [ZN*48]
                                float* __restrict__ Eout, int n) {
  int i = blockIdx.x * 256 + threadIdx.x;
  if (i >= n) return;
  int ek = i >> 7;           // edge index
  int d  = i & 127;
  int zn = ek / 48;
  Eout[i] = E[i] + a3[(size_t)zn * 128 + d] * bf2f(MSGE[i]) * mask[ek];
}

// ---------------------------------------------------------------------------
// Host-side GEMM dispatch (K is compile-time: 128 / 384 / 512)
// ---------------------------------------------------------------------------
template<int KK>
static void gemm_k(const unsigned short* A, const unsigned short* Bt,
                   const float* bias, void* C, int M, int N,
                   bool act, bool outbf, hipStream_t s) {
  dim3 g(N / 64, M / 16), b(32);
  size_t shmem = (size_t)64 * KK * 2;   // B panel in LDS
  if (act) {
    if (outbf) gemm_wmma_kernel<1, 1, KK><<<g, b, shmem, s>>>(A, Bt, bias, C, M, N);
    else       gemm_wmma_kernel<1, 0, KK><<<g, b, shmem, s>>>(A, Bt, bias, C, M, N);
  } else {
    if (outbf) gemm_wmma_kernel<0, 1, KK><<<g, b, shmem, s>>>(A, Bt, bias, C, M, N);
    else       gemm_wmma_kernel<0, 0, KK><<<g, b, shmem, s>>>(A, Bt, bias, C, M, N);
  }
}

static void launch_gemm(const unsigned short* A, const unsigned short* Bt,
                        const float* bias, void* C,
                        int M, int N, int K, bool act, bool outbf,
                        hipStream_t s) {
  if (K == 128)      gemm_k<128>(A, Bt, bias, C, M, N, act, outbf, s);
  else if (K == 384) gemm_k<384>(A, Bt, bias, C, M, N, act, outbf, s);
  else               gemm_k<512>(A, Bt, bias, C, M, N, act, outbf, s);
}

// ---------------------------------------------------------------------------
// Orchestration
// ---------------------------------------------------------------------------
extern "C" void kernel_launch(void* const* d_in, const int* in_sizes, int n_in,
                              void* d_out, int out_size, void* d_ws, size_t ws_size,
                              hipStream_t stream) {
  (void)in_sizes; (void)n_in; (void)out_size; (void)ws_size;
  constexpr int Zc = 2, Nn = 2048, KN = 48, D = 128;
  constexpr int ZN = Zc * Nn;        // 4096
  constexpr int M3 = ZN * KN;        // 196608

  const float* V    = (const float*)d_in[0];
  const float* E    = (const float*)d_in[1];
  const int*   Kx   = (const int*)d_in[2];
  const float* t    = (const float*)d_in[3];
  const float* mask = (const float*)d_in[4];

  struct WD { int idx, K, N; };
  static const WD wd[26] = {
    {5,384,128},{7,128,128},{9,128,128},          // node_msgr
    {11,128,384},{13,384,384},{15,384,768},       // nm_gb
    {17,128,128},{19,128,128},{21,128,128},       // nm_a
    {23,128,512},{25,512,128},                    // ffn
    {27,128,128},{29,128,128},{31,128,256},       // ffn_gb
    {33,128,128},{35,128,128},{37,128,128},       // ffn_a
    {39,384,128},{41,128,128},{43,128,128},       // edge_msgr
    {45,128,384},{47,384,384},{49,384,768},       // em_gb
    {51,128,128},{53,128,128},{55,128,128},       // em_a
  };

  char* p = (char*)d_ws;
  auto alloc = [&](size_t bytes) {
    char* r = p; p += (bytes + 255) & ~(size_t)255; return r;
  };

  unsigned short* t_bf = (unsigned short*)alloc((size_t)ZN * D * 2);
  unsigned short* Wt[26];
  for (int i = 0; i < 26; ++i)
    Wt[i] = (unsigned short*)alloc((size_t)wd[i].K * wd[i].N * 2);
  unsigned short* tmpA = (unsigned short*)alloc((size_t)ZN * 384 * 2);
  unsigned short* tmpB = (unsigned short*)alloc((size_t)ZN * 384 * 2);
  float* nm_gb_o  = (float*)alloc((size_t)ZN * 768 * 4);
  float* nm_a_o   = (float*)alloc((size_t)ZN * 128 * 4);
  float* ffn_gb_o = (float*)alloc((size_t)ZN * 256 * 4);
  float* ffn_a_o  = (float*)alloc((size_t)ZN * 128 * 4);
  float* em_gb_o  = (float*)alloc((size_t)ZN * 768 * 4);
  float* em_a_o   = (float*)alloc((size_t)ZN * 128 * 4);
  unsigned short* X1 = (unsigned short*)alloc((size_t)M3 * 384 * 2);
  unsigned short* H1 = (unsigned short*)alloc((size_t)M3 * 128 * 2);
  unsigned short* H2 = (unsigned short*)alloc((size_t)M3 * 128 * 2);
  float* Vln = (float*)alloc((size_t)ZN * 128 * 4);
  unsigned short* Vln_bf = (unsigned short*)alloc((size_t)ZN * 128 * 2);
  unsigned short* FH = (unsigned short*)alloc((size_t)ZN * 512 * 2);
  float* FO = (float*)alloc((size_t)ZN * 128 * 4);
  float* V2 = (float*)alloc((size_t)ZN * 128 * 4);

  // t -> bf16; weights -> transposed bf16
  f32_to_bf16_kernel<<<(ZN * D + 255) / 256, 256, 0, stream>>>(t, t_bf, ZN * D);
  for (int i = 0; i < 26; ++i) {
    int cnt = wd[i].K * wd[i].N;
    transpose_w_kernel<<<(cnt + 255) / 256, 256, 0, stream>>>(
        (const float*)d_in[wd[i].idx], Wt[i], wd[i].K, wd[i].N);
  }

  auto Bb = [&](int i) { return (const float*)d_in[wd[i].idx + 1]; };
  auto chain3 = [&](int a, int b, int c, float* out) {
    launch_gemm(t_bf, Wt[a], Bb(a), tmpA, ZN, wd[a].N, wd[a].K, true,  true,  stream);
    launch_gemm(tmpA, Wt[b], Bb(b), tmpB, ZN, wd[b].N, wd[b].K, true,  true,  stream);
    launch_gemm(tmpB, Wt[c], Bb(c), out,  ZN, wd[c].N, wd[c].K, false, false, stream);
  };
  chain3(3, 4, 5,  nm_gb_o);    // g1|b1
  chain3(6, 7, 8,  nm_a_o);     // a1
  chain3(11, 12, 13, ffn_gb_o); // g2|b2
  chain3(14, 15, 16, ffn_a_o);  // a2
  chain3(20, 21, 22, em_gb_o);  // g3|b3
  chain3(23, 24, 25, em_a_o);   // a3

  // ---- node message: X1 = g1*LN([Vi|Vj|E])+b1 ; msg = MLP(X1) ----
  prelude_kernel<<<M3, 128, 0, stream>>>(V, E, Kx, nm_gb_o, X1);
  launch_gemm(X1, Wt[0], Bb(0), H1, M3, 128, 384, true,  true, stream);
  launch_gemm(H1, Wt[1], Bb(1), H2, M3, 128, 128, true,  true, stream);
  launch_gemm(H2, Wt[2], Bb(2), H1, M3, 128, 128, false, true, stream); // MSG = H1
  node_update_kernel<<<ZN, 128, 0, stream>>>(H1, mask, nm_a_o, V, ffn_gb_o,
                                             Vln, Vln_bf);

  // ---- FFN on node features ----
  launch_gemm(Vln_bf, Wt[9],  Bb(9),  FH, ZN, 512, 128, true,  true,  stream);
  launch_gemm(FH,     Wt[10], Bb(10), FO, ZN, 128, 512, false, false, stream);
  float* Vout = (float*)d_out;
  v2_kernel<<<(ZN * D + 255) / 256, 256, 0, stream>>>(Vln, ffn_a_o, FO, V2,
                                                      Vout, ZN * D);

  // ---- edge message on updated V ----
  prelude_kernel<<<M3, 128, 0, stream>>>(V2, E, Kx, em_gb_o, X1);
  launch_gemm(X1, Wt[17], Bb(17), H1, M3, 128, 384, true,  true, stream);
  launch_gemm(H1, Wt[18], Bb(18), H2, M3, 128, 128, true,  true, stream);
  launch_gemm(H2, Wt[19], Bb(19), H1, M3, 128, 128, false, true, stream); // MSGE = H1
  float* Eout = (float*)d_out + (size_t)ZN * D;
  e_update_kernel<<<(M3 * D + 255) / 256, 256, 0, stream>>>(E, em_a_o, H1,
                                                            mask, Eout, M3 * D);
}
